// SelfAttention_71425306133046
// MI455X (gfx1250) — matrix operand order repaired and checked by reading, not compiled, not run on hardware
//
#include <hip/hip_runtime.h>

// CDNA5 / gfx1250 — wave32, WMMA bf16 16x16x32 with f32 accumulate.
typedef __attribute__((ext_vector_type(16))) __bf16 v16bf;
typedef __attribute__((ext_vector_type(8)))  __bf16 v8bf;
typedef __attribute__((ext_vector_type(8)))  float  v8f;

#define BB    8
#define CDIM  2048
#define DIMF  512
#define HDIM  512
#define NEG_SLOPE 0.01f

// ---------------------------------------------------------------------------
// Kernel 1: fold a_score into W_w / b_w:  wqa = W_w[:, :H] @ a[:H], etc.
// ---------------------------------------------------------------------------
__global__ void prep_kernel(const float* __restrict__ W_w,
                            const float* __restrict__ b_w,
                            const float* __restrict__ a_score,
                            float* __restrict__ wqa, float* __restrict__ wka,
                            float* __restrict__ bqk) {
    int d = blockIdx.x * blockDim.x + threadIdx.x;
    if (d < DIMF) {
        const float* row = W_w + (size_t)d * (3 * HDIM);
        float aq = 0.f, ak = 0.f;
        for (int h = 0; h < HDIM; ++h) {
            aq += row[h] * a_score[h];
            ak += row[HDIM + h] * a_score[HDIM + h];
        }
        wqa[d] = aq;
        wka[d] = ak;
    }
    if (blockIdx.x == 0 && threadIdx.x == 0) {
        float bq = 0.f, bk = 0.f;
        for (int h = 0; h < HDIM; ++h) {
            bq += b_w[h] * a_score[h];
            bk += b_w[HDIM + h] * a_score[HDIM + h];
        }
        bqk[0] = bq;
        bqk[1] = bk;
    }
}

// ---------------------------------------------------------------------------
// Kernel 2: sq[b,c] = features[b,c,:]·wqa + bq ; sk likewise. One wave per row.
// ---------------------------------------------------------------------------
__global__ void sqsk_kernel(const float* __restrict__ features,
                            const float* __restrict__ wqa,
                            const float* __restrict__ wka,
                            const float* __restrict__ bqk,
                            float* __restrict__ sq, float* __restrict__ sk) {
    int wave = (int)((blockIdx.x * blockDim.x + threadIdx.x) >> 5);
    int lane = threadIdx.x & 31;
    if (wave >= BB * CDIM) return;
    const float* row = features + (size_t)wave * DIMF;
    float aq = 0.f, ak = 0.f;
    for (int i = lane; i < DIMF; i += 32) {
        float f = row[i];
        aq += f * wqa[i];
        ak += f * wka[i];
    }
    #pragma unroll
    for (int off = 16; off > 0; off >>= 1) {
        aq += __shfl_xor(aq, off, 32);
        ak += __shfl_xor(ak, off, 32);
    }
    if (lane == 0) {
        sq[wave] = aq + bqk[0];
        sk[wave] = ak + bqk[1];
    }
}

// ---------------------------------------------------------------------------
// Kernel 3a: elementwise f32 -> bf16 (features), 8 elems/thread, b128 I/O.
// ---------------------------------------------------------------------------
__global__ void conv_f32_bf16_kernel(const float* __restrict__ src,
                                     __bf16* __restrict__ dst, int n8) {
    int i = blockIdx.x * blockDim.x + threadIdx.x;
    if (i >= n8) return;
    v8f v = *(const v8f*)(src + (size_t)i * 8);
    v8bf o;
    #pragma unroll
    for (int e = 0; e < 8; ++e) o[e] = (__bf16)v[e];
    *(v8bf*)(dst + (size_t)i * 8) = o;
}

// ---------------------------------------------------------------------------
// Kernel 3b: weight transpose + convert: Bt[n][k] = (bf16)B[k*ldb + n].
// One block per output row n; tiny (1MB), runs once.
// ---------------------------------------------------------------------------
__global__ void conv_wT_kernel(const float* __restrict__ B, int ldb,
                               __bf16* __restrict__ Bt, int K) {
    int n = blockIdx.x;
    for (int k = threadIdx.x; k < K; k += blockDim.x)
        Bt[(size_t)n * K + k] = (__bf16)B[(size_t)k * ldb + n];
}

// ---------------------------------------------------------------------------
// Shared bf16-WMMA GEMM core: A bf16 [M][K] row-major, Bt bf16 [N][K]
// (pre-transposed). One 16x16 tile per wave; all fragments contiguous loads.
// ---------------------------------------------------------------------------
__device__ __forceinline__ v8f gemm_core(const __bf16* __restrict__ A, int lda,
                                         const __bf16* __restrict__ Bt, int ldb,
                                         int row0, int col0, int K, int lane) {
    int hi = lane >> 4;
    int ln = lane & 15;
    const __bf16* Arow = A + (size_t)(row0 + ln) * lda + hi * 16;
    const __bf16* Brow = Bt + (size_t)(col0 + ln) * ldb + hi * 16;
    v8f acc = {0.f, 0.f, 0.f, 0.f, 0.f, 0.f, 0.f, 0.f};
    #pragma unroll 4
    for (int k = 0; k < K; k += 32) {
        v8bf alo = *(const v8bf*)(Arow + k);
        v8bf ahi = *(const v8bf*)(Arow + k + 8);
        v16bf a = __builtin_shufflevector(alo, ahi, 0, 1, 2, 3, 4, 5, 6, 7,
                                          8, 9, 10, 11, 12, 13, 14, 15);
        v16bf b = *(const v16bf*)(Brow + k);
        acc = __builtin_amdgcn_wmma_f32_16x16x32_bf16(false, a, false, b,
                                                      (short)0, acc, false, false);
    }
    return acc;
}

// ---------------------------------------------------------------------------
// Kernel 4: V = featbf @ WvT + bias, stored bf16 TRANSPOSED Vt[b][h][c]
// so flash-attention B-fragments are contiguous 32B loads.
// ---------------------------------------------------------------------------
__global__ void gemm_v_kernel(const __bf16* __restrict__ A, int lda,
                              const __bf16* __restrict__ Bt, int ldb,
                              const float* __restrict__ bias,
                              __bf16* __restrict__ Vt, int K) {
    int wave = threadIdx.x >> 5;
    int lane = threadIdx.x & 31;
    int hi   = lane >> 4;
    int ln   = lane & 15;
    int row0 = blockIdx.x * 16;
    int col0 = blockIdx.y * 128 + wave * 16;

    v8f acc = gemm_core(A, lda, Bt, ldb, row0, col0, K, lane);

    float bv = bias[col0 + ln];
    #pragma unroll
    for (int v = 0; v < 8; ++v) {
        int r  = row0 + v + 8 * hi;          // global row in [0, B*C)
        int b  = r >> 11;                    // / CDIM
        int cr = r & (CDIM - 1);
        Vt[((size_t)b * HDIM + col0 + ln) * CDIM + cr] = (__bf16)(acc[v] + bv);
    }
}

// ---------------------------------------------------------------------------
// Kernel 5: fused flash attention. Grid (C/16, B), block 256 (8 waves).
// Wave w owns H columns [64w, 64w+64) -> 4 accumulator tiles. Barrier-free:
// each wave redundantly builds the 16x32 P tile in WMMA A-layout registers.
// Output `weighted` emitted directly in bf16 (feeds the final bf16 GEMM).
// ---------------------------------------------------------------------------
__global__ void flash_attn_kernel(const float* __restrict__ sq,
                                  const float* __restrict__ sk,
                                  const float* __restrict__ mask,
                                  const __bf16* __restrict__ Vt,
                                  const float* __restrict__ b_score,
                                  __bf16* __restrict__ weighted) {
    int b    = blockIdx.y;
    int i0   = blockIdx.x * 16;
    int wave = threadIdx.x >> 5;
    int lane = threadIdx.x & 31;
    int hi   = lane >> 4;
    int ln   = lane & 15;

    float         b0      = b_score[0];
    float         sqr     = sq[(size_t)b * CDIM + i0 + ln];     // my row's sq
    const float*  sks     = sk + (size_t)b * CDIM;
    const float*  maskrow = mask + ((size_t)b * CDIM + i0 + ln) * CDIM;
    const __bf16* Vtb     = Vt + (size_t)b * HDIM * CDIM;
    int hbase = wave * 64;

    float m_r = -1e30f, l_r = 0.f;
    v8f acc[4];
    #pragma unroll
    for (int t = 0; t < 4; ++t)
        acc[t] = (v8f){0.f, 0.f, 0.f, 0.f, 0.f, 0.f, 0.f, 0.f};

    for (int j0 = 0; j0 < CDIM; j0 += 32) {
        // mask is the one true HBM stream: prefetch 8 tiles ahead
        if (j0 + 256 < CDIM)
            __builtin_prefetch(maskrow + j0 + 256, 0, 3);

        // ---- scores for row ln, keys [c,c+16), c=16*hi
        int   c = 16 * hi;
        float s[16];
        float mx = -1e30f;
        #pragma unroll
        for (int e = 0; e < 16; ++e) {
            int   j  = j0 + c + e;
            float sv = sqr + sks[j] + b0;
            sv = (sv > 0.f) ? sv : sv * NEG_SLOPE;   // leaky_relu
            sv *= maskrow[j];                        // mask BEFORE softmax
            s[e] = sv;
            mx = fmaxf(mx, sv);
        }
        mx = fmaxf(mx, __shfl_xor(mx, 16, 32));      // join the two lane halves
        float m_new = fmaxf(m_r, mx);
        float alpha = __expf(m_r - m_new);

        float psum = 0.f;
        v16bf p;
        #pragma unroll
        for (int e = 0; e < 16; ++e) {
            float pe = __expf(s[e] - m_new);
            psum += pe;
            p[e] = (__bf16)pe;
        }
        psum += __shfl_xor(psum, 16, 32);
        l_r = l_r * alpha + psum;
        m_r = m_new;

        // rescale accumulators: C-tile element v is row (v + 8*hi)
        float arow[8];
        #pragma unroll
        for (int v = 0; v < 8; ++v) arow[v] = __shfl(alpha, v + 8 * hi, 32);
        #pragma unroll
        for (int t = 0; t < 4; ++t)
            #pragma unroll
            for (int v = 0; v < 8; ++v) acc[t][v] *= arow[v];

        // ---- P(16x32) @ V(32x16) per H tile; Vt is bf16 [H][C] contiguous
        int kb = 16 * hi;
        #pragma unroll
        for (int t = 0; t < 4; ++t) {
            int h = hbase + t * 16 + ln;
            v16bf bfr = *(const v16bf*)(Vtb + (size_t)h * CDIM + j0 + kb);
            acc[t] = __builtin_amdgcn_wmma_f32_16x16x32_bf16(false, p, false, bfr,
                                                             (short)0, acc[t],
                                                             false, false);
        }
    }

    // normalize by row sums and store weighted [B][C][H] (bf16)
    float lrow[8];
    #pragma unroll
    for (int v = 0; v < 8; ++v) lrow[v] = __shfl(l_r, v + 8 * hi, 32);
    #pragma unroll
    for (int t = 0; t < 4; ++t) {
        #pragma unroll
        for (int v = 0; v < 8; ++v) {
            int r = i0 + v + 8 * hi;
            weighted[((size_t)b * CDIM + r) * HDIM + hbase + t * 16 + ln] =
                (__bf16)(acc[t][v] / lrow[v]);
        }
    }
}

// ---------------------------------------------------------------------------
// Kernel 6: out = weighted @ W_out + b_out  (bf16 in, f32 out)
// ---------------------------------------------------------------------------
__global__ void gemm_out_kernel(const __bf16* __restrict__ A, int lda,
                                const __bf16* __restrict__ Bt, int ldb,
                                const float* __restrict__ bias,
                                float* __restrict__ C, int ldc, int K) {
    int wave = threadIdx.x >> 5;
    int lane = threadIdx.x & 31;
    int hi   = lane >> 4;
    int ln   = lane & 15;
    int row0 = blockIdx.x * 16;
    int col0 = blockIdx.y * 128 + wave * 16;

    v8f acc = gemm_core(A, lda, Bt, ldb, row0, col0, K, lane);

    float bv = bias[col0 + ln];
    #pragma unroll
    for (int v = 0; v < 8; ++v) {
        int r = row0 + v + 8 * hi;
        C[(size_t)r * ldc + col0 + ln] = acc[v] + bv;
    }
}

// ---------------------------------------------------------------------------
extern "C" void kernel_launch(void* const* d_in, const int* in_sizes, int n_in,
                              void* d_out, int out_size, void* d_ws, size_t ws_size,
                              hipStream_t stream) {
    const float* features = (const float*)d_in[0];   // [B,C,DIM]
    const float* mask     = (const float*)d_in[1];   // [B,C,C]
    const float* W_w      = (const float*)d_in[2];   // [DIM,3H]
    const float* b_w      = (const float*)d_in[3];   // [3H]
    const float* a_score  = (const float*)d_in[4];   // [2H]
    const float* b_score  = (const float*)d_in[5];   // [1]
    const float* W_out    = (const float*)d_in[6];   // [H,DIM]
    const float* b_out    = (const float*)d_in[7];   // [DIM]
    float* out = (float*)d_out;

    // Workspace layout (float units):
    //   wqa[512] | wka[512] | bqk | sq[16384] | sk[16384] | pad ->
    //   featbf bf16 [B*C][DIM]      @  34816  (4194304 f)
    //   Vt     bf16 [B][H][C]       @ 4229120 (4194304 f)
    //   WvT    bf16 [N=512][K=512]  @ 8423424 ( 131072 f)
    //   WoT    bf16 [N=512][K=512]  @ 8554496 ( 131072 f)
    //   wtd    bf16 [B][C][H]       @ 8685568 (4194304 f)   total ~51.5 MB
    float*  ws     = (float*)d_ws;
    float*  wqa    = ws;
    float*  wka    = ws + 512;
    float*  bqk    = ws + 1024;
    float*  sqv    = ws + 1536;
    float*  skv    = ws + 1536 + BB * CDIM;
    __bf16* featbf = (__bf16*)(ws + 34816);
    __bf16* Vt     = (__bf16*)(ws + 4229120);
    __bf16* WvT    = (__bf16*)(ws + 8423424);
    __bf16* WoT    = (__bf16*)(ws + 8554496);
    __bf16* wtd    = (__bf16*)(ws + 8685568);

    const int ROWS = BB * CDIM;   // 16384

    prep_kernel<<<2, 256, 0, stream>>>(W_w, b_w, a_score, wqa, wka, bqk);

    sqsk_kernel<<<ROWS / 8, 256, 0, stream>>>(features, wqa, wka, bqk, sqv, skv);

    // one-time operand conversions (bf16 / transposed weights)
    conv_f32_bf16_kernel<<<(ROWS * DIMF / 8 + 255) / 256, 256, 0, stream>>>(
        features, featbf, ROWS * DIMF / 8);
    conv_wT_kernel<<<HDIM, 256, 0, stream>>>(W_w + 2 * HDIM, 3 * HDIM, WvT, DIMF);
    conv_wT_kernel<<<DIMF, 256, 0, stream>>>(W_out, DIMF, WoT, HDIM);

    dim3 gGemm(ROWS / 16, DIMF / 128);
    gemm_v_kernel<<<gGemm, 256, 0, stream>>>(featbf, DIMF, WvT, DIMF,
                                             b_w + 2 * HDIM, Vt, DIMF);

    dim3 gAttn(CDIM / 16, BB);
    flash_attn_kernel<<<gAttn, 256, 0, stream>>>(sqv, skv, mask, Vt, b_score, wtd);

    gemm_out_kernel<<<gGemm, 256, 0, stream>>>(wtd, HDIM, WoT, HDIM, b_out,
                                               out, DIMF, HDIM);
}
